// SqueezeBertSelfAttention_42004780155203
// MI455X (gfx1250) — matrix operand order repaired and checked by reading, not compiled, not run on hardware
//
#include <hip/hip_runtime.h>

// ---------------------------------------------------------------------------
// SqueezeBert self-attention for MI455X (gfx1250, wave32, WMMA bf16 path)
// N=8, C=1024, W=1024, H=16, D=64
// ---------------------------------------------------------------------------

typedef __bf16 v8bf  __attribute__((ext_vector_type(8)));
typedef __bf16 v16bf __attribute__((ext_vector_type(16)));
typedef float  v8f   __attribute__((ext_vector_type(8)));

#define WMMA_BF16(A, B, Cacc) \
    __builtin_amdgcn_wmma_f32_16x16x32_bf16(false, (A), false, (B), (short)0, (Cacc), false, false)

static constexpr int NB = 8;
static constexpr int C  = 1024;
static constexpr int W  = 1024;
static constexpr int H  = 16;
static constexpr int D  = 64;

__device__ __forceinline__ v16bf cat16(v8bf lo, v8bf hi) {
    return __builtin_shufflevector(lo, hi, 0,1,2,3,4,5,6,7,8,9,10,11,12,13,14,15);
}

__device__ __forceinline__ v8bf pack8(float4 a, float4 b) {
    v8bf r;
    r[0] = (__bf16)a.x; r[1] = (__bf16)a.y; r[2] = (__bf16)a.z; r[3] = (__bf16)a.w;
    r[4] = (__bf16)b.x; r[5] = (__bf16)b.y; r[6] = (__bf16)b.z; r[7] = (__bf16)b.w;
    return r;
}

// ---------------------------------------------------------------------------
// Kernel 1: QKV projection GEMM.  out[o,w] = sum_c Wm[o,c]*hs[n,c,w] + b[o]
// grid = (W/64, C/64, NB*3), block = 128 (4 waves). 64x64 tile, K-step 32.
// which==0 -> Q written as [N][H][W][D] bf16 (q-transposed for attention A-frags)
// which==1 -> K written as [N][H][W][D] bf16 (so K B-frags are d-contiguous)
// which==2 -> V written as [N][C][W]    bf16 (so V B-frags are w-contiguous)
// ---------------------------------------------------------------------------
__global__ void __launch_bounds__(128)
qkv_gemm(const float* __restrict__ hs,
         const float* __restrict__ wqm, const float* __restrict__ wkm,
         const float* __restrict__ wvm,
         const float* __restrict__ bq,  const float* __restrict__ bk,
         const float* __restrict__ bv,
         __bf16* __restrict__ qT, __bf16* __restrict__ kT,
         __bf16* __restrict__ vn)
{
    const int w0    = blockIdx.x * 64;
    const int m0    = blockIdx.y * 64;
    const int n     = blockIdx.z / 3;
    const int which = blockIdx.z % 3;

    const float* Wmat = (which == 0) ? wqm : (which == 1) ? wkm : wvm;
    const float* bias = (which == 0) ? bq  : (which == 1) ? bk  : bv;
    const float* hsn  = hs + (size_t)n * C * W;

    constexpr int AP = 40;   // LDS pitch (32 K + 8 pad) elements
    constexpr int BP = 40;
    __shared__ __bf16 lds_a[64 * AP];   // A tile  [m][k]
    __shared__ __bf16 lds_b[64 * BP];   // B tile transposed [w][k]

    const int t    = threadIdx.x;       // 0..127
    const int wid  = t >> 5;            // wave 0..3 -> M strip
    const int lane = t & 31;
    const int g    = lane >> 4;         // half-wave group
    const int lx   = lane & 15;

    v8f acc[4] = {};                    // 4 N-blocks of 16x16 f32

    for (int k0 = 0; k0 < C; k0 += 32) {
        // ---- stage A: weight rows m0..m0+63, cols k0..k0+31 (f32 -> bf16)
        {
            const int row  = t >> 1;
            const int half = t & 1;
            const float4* gp = (const float4*)(Wmat + (size_t)(m0 + row) * C + k0 + half * 16);
            float4 f0 = gp[0], f1 = gp[1], f2 = gp[2], f3 = gp[3];
            *(v8bf*)(lds_a + row * AP + half * 16)     = pack8(f0, f1);
            *(v8bf*)(lds_a + row * AP + half * 16 + 8) = pack8(f2, f3);
            if (k0 + 32 < C)
                __builtin_prefetch(Wmat + (size_t)(m0 + row) * C + k0 + 32 + half * 16, 0, 0);
        }
        // ---- stage B (transposed): hs rows k0..k0+31, cols w0..w0+63
        {
            const int kr   = t >> 2;
            const int quad = t & 3;
            const float4* gp = (const float4*)(hsn + (size_t)(k0 + kr) * W + w0 + quad * 16);
            float4 f0 = gp[0], f1 = gp[1], f2 = gp[2], f3 = gp[3];
            v8bf x0 = pack8(f0, f1), x1 = pack8(f2, f3);
            #pragma unroll
            for (int e = 0; e < 8; ++e) {
                lds_b[(quad * 16 + e)     * BP + kr] = x0[e];
                lds_b[(quad * 16 + 8 + e) * BP + kr] = x1[e];
            }
            if (k0 + 32 < C)
                __builtin_prefetch(hsn + (size_t)(k0 + 32 + kr) * W + w0 + quad * 16, 0, 0);
        }
        __syncthreads();

        // ---- A fragment for this wave's 16-row M strip
        const __bf16* ar = lds_a + (wid * 16 + lx) * AP;
        v16bf afrag = cat16(*(const v8bf*)(ar + g * 16),
                            *(const v8bf*)(ar + g * 16 + 8));
        #pragma unroll
        for (int nb = 0; nb < 4; ++nb) {
            const __bf16* br = lds_b + (nb * 16 + lx) * BP + g * 16;
            v16bf bfrag = cat16(*(const v8bf*)br, *(const v8bf*)(br + 8));
            acc[nb] = WMMA_BF16(afrag, bfrag, acc[nb]);
        }
        __syncthreads();
    }

    // ---- epilogue: bias add, bf16 convert, layout-specific store
    #pragma unroll
    for (int nb = 0; nb < 4; ++nb) {
        const int wcol = w0 + nb * 16 + lx;
        #pragma unroll
        for (int r = 0; r < 8; ++r) {
            const int o   = m0 + wid * 16 + r + 8 * g;   // output channel
            float  val    = acc[nb][r] + bias[o];
            __bf16 bval   = (__bf16)val;
            if (which == 2) {
                vn[((size_t)n * C + o) * W + wcol] = bval;
            } else {
                const int hh = o >> 6, dd = o & 63;
                __bf16* dst = (which == 0) ? qT : kT;
                dst[(((size_t)n * H + hh) * W + wcol) * D + dd] = bval;
            }
        }
    }
}

// ---------------------------------------------------------------------------
// Kernel 2: flash attention, one wave per 16 query rows of one (n,h).
// grid = (W/16, H, NB), block = 32.
// S = Q Kt / 8 + mask ; online softmax ; O += P V ; out = O / l
// ---------------------------------------------------------------------------
__global__ void __launch_bounds__(32)
attn_flash(const __bf16* __restrict__ qT,   // [N][H][W][D]
           const __bf16* __restrict__ kT,   // [N][H][W][D]
           const __bf16* __restrict__ vn,   // [N][C][W]
           const float*  __restrict__ amask,// [N][1][1][W]
           float* __restrict__ out)         // [N][C][W]
{
    const int q0   = blockIdx.x * 16;
    const int h    = blockIdx.y;
    const int n    = blockIdx.z;
    const int lane = threadIdx.x & 31;
    const int g    = lane >> 4;
    const int lx   = lane & 15;

    constexpr int PP = 40;                  // P tile pitch (32 cols + 8 pad)
    __shared__ __bf16 p_lds[16 * PP];

    const float   scale = 0.125f;           // 1/sqrt(64)
    const __bf16* qb = qT + (((size_t)n * H + h) * W) * D;
    const __bf16* kb = kT + (((size_t)n * H + h) * W) * D;
    const __bf16* vb = vn + ((size_t)n * C + h * D) * W;
    const float*  mk = amask + (size_t)n * W;

    // Q A-fragments (row = q0+lx), two 32-wide d-chunks, loaded once
    v16bf qf[2];
    #pragma unroll
    for (int dc = 0; dc < 2; ++dc) {
        const __bf16* p = qb + (size_t)(q0 + lx) * D + dc * 32 + g * 16;
        qf[dc] = cat16(*(const v8bf*)p, *(const v8bf*)(p + 8));
    }

    float m_i[8], l_i[8];
    #pragma unroll
    for (int r = 0; r < 8; ++r) { m_i[r] = -3.0e38f; l_i[r] = 0.0f; }
    v8f o_acc[4] = {};

    for (int kk = 0; kk < W; kk += 32) {
        // ---- S block: 16 queries x 32 keys
        v8f s[2] = {};
        #pragma unroll
        for (int cb = 0; cb < 2; ++cb) {
            #pragma unroll
            for (int dc = 0; dc < 2; ++dc) {
                const __bf16* p = kb + (size_t)(kk + cb * 16 + lx) * D + dc * 32 + g * 16;
                v16bf kf = cat16(*(const v8bf*)p, *(const v8bf*)(p + 8));
                s[cb] = WMMA_BF16(qf[dc], kf, s[cb]);
            }
        }
        const float mv0 = mk[kk + lx];
        const float mv1 = mk[kk + 16 + lx];

        // ---- online softmax (rows r+8g live in lane-group g)
        #pragma unroll
        for (int r = 0; r < 8; ++r) {
            float s0 = s[0][r] * scale + mv0;
            float s1 = s[1][r] * scale + mv1;
            float bm = fmaxf(s0, s1);
            #pragma unroll
            for (int off = 8; off >= 1; off >>= 1)
                bm = fmaxf(bm, __shfl_xor(bm, off, 16));
            const float mnew  = fmaxf(m_i[r], bm);
            const float alpha = __expf(m_i[r] - mnew);
            const float p0 = __expf(s0 - mnew);
            const float p1 = __expf(s1 - mnew);
            float rs = p0 + p1;
            #pragma unroll
            for (int off = 8; off >= 1; off >>= 1)
                rs += __shfl_xor(rs, off, 16);
            l_i[r] = l_i[r] * alpha + rs;
            m_i[r] = mnew;
            #pragma unroll
            for (int b = 0; b < 4; ++b) o_acc[b][r] *= alpha;
            p_lds[(r + 8 * g) * PP + lx]      = (__bf16)p0;
            p_lds[(r + 8 * g) * PP + 16 + lx] = (__bf16)p1;
        }
        asm volatile("s_wait_dscnt 0" ::: "memory");   // single wave: LDS RAW fence

        // ---- P A-fragment (C-layout -> A-layout via LDS)
        const __bf16* pr = p_lds + lx * PP;
        v16bf pf = cat16(*(const v8bf*)(pr + g * 16),
                         *(const v8bf*)(pr + g * 16 + 8));

        // ---- O += P * V   (V B-frags contiguous in key dim)
        #pragma unroll
        for (int b = 0; b < 4; ++b) {
            const __bf16* vp = vb + (size_t)(b * 16 + lx) * W + kk + g * 16;
            v16bf vf = cat16(*(const v8bf*)vp, *(const v8bf*)(vp + 8));
            o_acc[b] = WMMA_BF16(pf, vf, o_acc[b]);
        }
    }

    // ---- normalize and store:  out[n][h*64+d][q]
    #pragma unroll
    for (int b = 0; b < 4; ++b) {
        const int d = b * 16 + lx;
        float* ob = out + ((size_t)n * C + h * D + d) * W + q0;
        #pragma unroll
        for (int r = 0; r < 8; ++r)
            ob[r + 8 * g] = o_acc[b][r] / l_i[r];
    }
}

// ---------------------------------------------------------------------------
// Launch: inputs in setup_inputs() order:
//   0 hidden_states [N,C,W] f32   1 attention_mask [N,1,1,W] f32
//   2 wq [C,C]  3 bq [C]  4 wk  5 bk  6 wv  7 bv
// Workspace: qT (16MB) | kT (16MB) | vn (16MB)  = 48 MB bf16
// ---------------------------------------------------------------------------
extern "C" void kernel_launch(void* const* d_in, const int* in_sizes, int n_in,
                              void* d_out, int out_size, void* d_ws, size_t ws_size,
                              hipStream_t stream)
{
    const float* hs   = (const float*)d_in[0];
    const float* mask = (const float*)d_in[1];
    const float* wq   = (const float*)d_in[2];
    const float* bq   = (const float*)d_in[3];
    const float* wk   = (const float*)d_in[4];
    const float* bk   = (const float*)d_in[5];
    const float* wv   = (const float*)d_in[6];
    const float* bv   = (const float*)d_in[7];
    float* out = (float*)d_out;

    const size_t M = (size_t)1 << 20;        // 1M elements
    __bf16* ws = (__bf16*)d_ws;
    __bf16* qT = ws;                          // 8M elems
    __bf16* kT = ws + 8 * M;                  // 8M elems
    __bf16* vn = ws + 16 * M;                 // 8M elems

    dim3 ggrid(W / 64, C / 64, NB * 3);       // 16 x 16 x 24
    qkv_gemm<<<ggrid, 128, 0, stream>>>(hs, wq, wk, wv, bq, bk, bv, qT, kT, vn);

    dim3 agrid(W / 16, H, NB);                // 64 x 16 x 8
    attn_flash<<<agrid, 32, 0, stream>>>(qT, kT, vn, mask, out);
}
